// SetCriterion_26233660244784
// MI455X (gfx1250) — compile-verified, matrix operand order checked by reading, not run on hardware
//
#include <hip/hip_runtime.h>
#include <math.h>

// ---------------- problem constants (match reference) ----------------
#define Bb   4
#define Nn   500
#define Cc   80
#define Tt   28
#define TTe  784          // T*T
#define Mm   16
#define Hh   800
#define Ww   800
#define KK   10           // SimOTA candidate top-k
#define NTIL 32           // ceil(500/16)

#define ALPHA_  0.25f
#define GAMMA_  2.0f
#define EPSF    1e-8f
#define CTR_R   0.25f
#define CLS_W   2.0f
#define L1_W    5.0f
#define GIOU_W  2.0f
#define MASK_W  5.0f

typedef __attribute__((ext_vector_type(16))) _Float16 v16h;
typedef __attribute__((ext_vector_type(8)))  _Float16 v8h;
typedef __attribute__((ext_vector_type(8)))  float    v8f;

__device__ __forceinline__ float sigf(float x) { return 1.0f / (1.0f + __expf(-x)); }
// fast sigmoid for the f16 WMMA path: v_exp_f32 + v_rcp_f32 only
__device__ __forceinline__ float sigfast(float x) {
  return __builtin_amdgcn_rcpf(1.0f + __expf(-x));
}

__device__ __forceinline__ float d_iou(float ax1,float ay1,float ax2,float ay2,
                                       float bx1,float by1,float bx2,float by2) {
  float iw = fmaxf(fminf(ax2,bx2) - fmaxf(ax1,bx1), 0.0f);
  float ih = fmaxf(fminf(ay2,by2) - fmaxf(ay1,by1), 0.0f);
  float in = iw * ih;
  float a1 = (ax2-ax1)*(ay2-ay1), a2 = (bx2-bx1)*(by2-by1);
  return in / (a1 + a2 - in + EPSF);
}

__device__ __forceinline__ float d_giou(float ax1,float ay1,float ax2,float ay2,
                                        float bx1,float by1,float bx2,float by2) {
  float iw = fmaxf(fminf(ax2,bx2) - fmaxf(ax1,bx1), 0.0f);
  float ih = fmaxf(fminf(ay2,by2) - fmaxf(ay1,by1), 0.0f);
  float in = iw * ih;
  float a1 = (ax2-ax1)*(ay2-ay1), a2 = (bx2-bx1)*(by2-by1);
  float un = a1 + a2 - in;
  float iou = in / (un + EPSF);
  float ew = fmaxf(fmaxf(ax2,bx2) - fminf(ax1,bx1), 0.0f);
  float eh = fmaxf(fmaxf(ay2,by2) - fminf(ay1,by1), 0.0f);
  float enc = ew * eh;
  return iou - (enc - un) / (enc + EPSF);
}

// ---------------- kernel 0: zero atomic accumulators ----------------
__global__ void k_init(float* gmsum, float* accum) {
  int i = threadIdx.x;
  if (i < Bb*Mm) gmsum[i] = 0.0f;
  if (i < 8)     accum[i] = 0.0f;
}

// ---------------- kernel 1: ROIAlign-style crop+binarize of GT masks -----
__global__ void k_crop(const float* __restrict__ gt_boxes,
                       const float* __restrict__ gt_masks,
                       _Float16* __restrict__ gmh,
                       float* __restrict__ gmsum) {
  int bm = blockIdx.x;           // b*Mm + m
  int t  = threadIdx.x;          // 0..783
  if (t >= TTe) return;
  int ty = t / Tt, tx = t % Tt;
  const float* bx = gt_boxes + bm * 4;
  float x1 = bx[0], y1 = bx[1], x2 = bx[2], y2 = bx[3];
  float x = x1 + (tx + 0.5f) * (x2 - x1) / (float)Tt - 0.5f;
  float y = y1 + (ty + 0.5f) * (y2 - y1) / (float)Tt - 0.5f;
  int x0 = (int)floorf(x), y0 = (int)floorf(y);
  float fx = x - (float)x0, fy = y - (float)y0;
  const float* mk = gt_masks + (size_t)bm * Hh * Ww;
  auto samp = [&](int iy, int ix) -> float {
    return (iy >= 0 && iy < Hh && ix >= 0 && ix < Ww) ? mk[(size_t)iy * Ww + ix] : 0.0f;
  };
  float s = samp(y0,   x0  ) * (1.0f-fy) * (1.0f-fx)
          + samp(y0,   x0+1) * (1.0f-fy) * fx
          + samp(y0+1, x0  ) * fy        * (1.0f-fx)
          + samp(y0+1, x0+1) * fy        * fx;
  float v = (s >= 0.5f) ? 1.0f : 0.0f;
  gmh[(size_t)bm * TTe + t] = (_Float16)v;
  atomicAdd(&gmsum[bm], v);
}

// ---------------- kernel 2: WMMA dice intersection + row sums ----------
// One wave per (b, n-tile of 16, m). Channel c = gt_cls[b,m] is gathered,
// so only 16/80 channels of pred_masks are read (~100MB vs ~502MB -> ~4.3us
// HBM floor at 23.3 TB/s). 24 chunks are statically fully in-range (both
// 8-element K segments), so the hot loop has zero predication and pure
// 16-byte vector loads; only the final chunk (k0=768) zero-pads segment 2.
// Tail rows n>=500 only affect unstored output rows, so the row pointer is
// merely clamped — no masking math.
__global__ __launch_bounds__(32)
void k_wmma(const float* __restrict__ pred_masks,
            const int*   __restrict__ gt_cls,
            const _Float16* __restrict__ gmh,
            float* __restrict__ inter,
            float* __restrict__ rowsum) {
  int bid   = blockIdx.x;
  int m     = bid % Mm;
  int ntile = (bid / Mm) % NTIL;
  int b     = bid / (Mm * NTIL);
  int lane  = threadIdx.x;
  int lo    = lane & 15;
  int kh    = lane >> 4;

  int c  = gt_cls[b * Mm + m];
  int n  = ntile * 16 + lo;
  int nc = (n < Nn) ? n : (Nn - 1);          // clamp for address safety only
  // per-lane segment bases (already include the +8*kh lane offset)
  const float*    Ab = pred_masks + ((size_t)(b * Nn + nc) * Cc + c) * TTe + 8 * kh;
  const _Float16* Gb = gmh + (size_t)(b * Mm + lo) * TTe + 8 * kh;

  v8f accg = {};
  v8f acc1 = {};
  v16h ones;
#pragma unroll
  for (int e = 0; e < 16; ++e) ones[e] = (_Float16)1.0f;

  // ---- 24 full chunks: everything statically in range ----
  for (int k0 = 0; k0 < TTe - 32; k0 += 32) {
    __builtin_prefetch(Ab + k0 + 32, 0, 1);        // next chunk, stays in-row
    float4 a0 = *(const float4*)(Ab + k0);         // K seg1: e=0..3
    float4 a1 = *(const float4*)(Ab + k0 + 4);     //         e=4..7
    float4 a2 = *(const float4*)(Ab + k0 + 16);    // K seg2: e=8..11
    float4 a3 = *(const float4*)(Ab + k0 + 20);    //         e=12..15
    v8h g0 = *(const v8h*)(Gb + k0);
    v8h g1 = *(const v8h*)(Gb + k0 + 16);
    v16h a, bf;
    a[0] = (_Float16)sigfast(a0.x);  a[1] = (_Float16)sigfast(a0.y);
    a[2] = (_Float16)sigfast(a0.z);  a[3] = (_Float16)sigfast(a0.w);
    a[4] = (_Float16)sigfast(a1.x);  a[5] = (_Float16)sigfast(a1.y);
    a[6] = (_Float16)sigfast(a1.z);  a[7] = (_Float16)sigfast(a1.w);
    a[8] = (_Float16)sigfast(a2.x);  a[9] = (_Float16)sigfast(a2.y);
    a[10] = (_Float16)sigfast(a2.z); a[11] = (_Float16)sigfast(a2.w);
    a[12] = (_Float16)sigfast(a3.x); a[13] = (_Float16)sigfast(a3.y);
    a[14] = (_Float16)sigfast(a3.z); a[15] = (_Float16)sigfast(a3.w);
#pragma unroll
    for (int e = 0; e < 8; ++e) { bf[e] = g0[e]; bf[8 + e] = g1[e]; }
    accg = __builtin_amdgcn_wmma_f32_16x16x32_f16(false, a, false, bf,   (short)0, accg, false, false);
    acc1 = __builtin_amdgcn_wmma_f32_16x16x32_f16(false, a, false, ones, (short)0, acc1, false, false);
  }

  // ---- last chunk (k0 = 768): segment 1 fully valid, segment 2 fully OOB ----
  {
    const int k0 = TTe - 16;                       // 768
    float4 a0 = *(const float4*)(Ab + k0);
    float4 a1 = *(const float4*)(Ab + k0 + 4);
    v8h g0 = *(const v8h*)(Gb + k0);
    v16h a, bf;
    a[0] = (_Float16)sigfast(a0.x);  a[1] = (_Float16)sigfast(a0.y);
    a[2] = (_Float16)sigfast(a0.z);  a[3] = (_Float16)sigfast(a0.w);
    a[4] = (_Float16)sigfast(a1.x);  a[5] = (_Float16)sigfast(a1.y);
    a[6] = (_Float16)sigfast(a1.z);  a[7] = (_Float16)sigfast(a1.w);
#pragma unroll
    for (int e = 0; e < 8; ++e) {
      a[8 + e]  = (_Float16)0.0f;
      bf[e]     = g0[e];
      bf[8 + e] = (_Float16)0.0f;
    }
    accg = __builtin_amdgcn_wmma_f32_16x16x32_f16(false, a, false, bf,   (short)0, accg, false, false);
    acc1 = __builtin_amdgcn_wmma_f32_16x16x32_f16(false, a, false, ones, (short)0, acc1, false, false);
  }

  // C/D layout: lane holds N = lane%16, rows M = v + 8*(lane>>4). Keep N == m.
  if (lo == m) {
#pragma unroll
    for (int v = 0; v < 8; ++v) {
      int nr = ntile * 16 + v + 8 * kh;
      if (nr < Nn) {
        size_t o = (size_t)(b * Nn + nr) * Mm + m;
        inter[o]  = accg[v];
        rowsum[o] = acc1[v];
      }
    }
  }
}

// ---------------- kernel 3: cost matrix + gated IoU ----------------
__global__ void k_cost(const float* __restrict__ logits,
                       const float* __restrict__ boxes,
                       const int*   __restrict__ gt_cls,
                       const float* __restrict__ gt_boxes,
                       const float* __restrict__ image_size,
                       const float* __restrict__ inter,
                       const float* __restrict__ rowsum,
                       const float* __restrict__ gmsum,
                       float* __restrict__ cost,
                       float* __restrict__ iouv,
                       float* __restrict__ validf) {
  int tid = blockIdx.x * blockDim.x + threadIdx.x;
  if (tid >= Bb * Nn) return;
  int b = tid / Nn;
  const float* bx  = boxes + (size_t)tid * 4;
  const float* img = image_size + b * 4;
  float nb0 = bx[0]/img[0], nb1 = bx[1]/img[1], nb2 = bx[2]/img[2], nb3 = bx[3]/img[3];
  float cx = (bx[0] + bx[2]) * 0.5f, cy = (bx[1] + bx[3]) * 0.5f;

  unsigned inboth = 0u;
  bool valid = false;
  float iou_l[Mm];
  for (int m = 0; m < Mm; ++m) {
    const float* g = gt_boxes + (size_t)(b * Mm + m) * 4;
    bool ib = (cx > g[0]) && (cx < g[2]) && (cy > g[1]) && (cy < g[3]);
    float gcx = (g[0]+g[2])*0.5f, gcy = (g[1]+g[3])*0.5f;
    float gw = g[2]-g[0], gh = g[3]-g[1];
    bool ic = (fabsf(cx-gcx) < CTR_R*gw) && (fabsf(cy-gcy) < CTR_R*gh);
    if (ib && ic) inboth |= (1u << m);
    valid = valid || ib || ic;
    iou_l[m] = d_iou(bx[0],bx[1],bx[2],bx[3], g[0],g[1],g[2],g[3]);
  }

  for (int m = 0; m < Mm; ++m) {
    int c = gt_cls[b * Mm + m];
    float l  = logits[(size_t)tid * Cc + c];
    float pc = sigf(l);
    float pos = -logf(pc + EPSF)        * ALPHA_        * (1.0f-pc) * (1.0f-pc);
    float neg = -logf(1.0f - pc + EPSF) * (1.0f-ALPHA_) * pc * pc;
    float cls_c = CLS_W * (pos - neg);

    const float* g = gt_boxes + (size_t)(b * Mm + m) * 4;
    float ng0 = g[0]/img[0], ng1 = g[1]/img[1], ng2 = g[2]/img[2], ng3 = g[3]/img[3];
    float l1_c = L1_W * (fabsf(nb0-ng0)+fabsf(nb1-ng1)+fabsf(nb2-ng2)+fabsf(nb3-ng3));
    float gi_c = GIOU_W * (1.0f - d_giou(nb0,nb1,nb2,nb3, ng0,ng1,ng2,ng3));

    size_t o = (size_t)tid * Mm + m;
    float mu = rowsum[o] + gmsum[b * Mm + m] + EPSF;
    float mk_c = MASK_W * (1.0f - 2.0f * inter[o] / mu);

    float cst = cls_c + l1_c + gi_c + mk_c;
    if (!((inboth >> m) & 1u)) cst += 1e5f;
    if (!valid)                cst += 1e9f;
    cost[o] = cst;
    iouv[o] = valid ? iou_l[m] : 0.0f;
  }
  validf[tid] = valid ? 1.0f : 0.0f;
}

// ---------------- kernel 4: per-GT-column dynamic-k + candidate top-10 -----
__global__ void k_cols(const float* __restrict__ iouv,
                       const float* __restrict__ cost,
                       int* __restrict__ idx_sel,
                       int* __restrict__ dynk) {
  int t = threadIdx.x;
  if (t >= Bb * Mm) return;
  int b = t / Mm, m = t % Mm;
  const float* ic = iouv + (size_t)b * Nn * Mm + m;   // stride Mm
  const float* cc = cost + (size_t)b * Nn * Mm + m;
  int sel[KK];

  float s = 0.0f;
  for (int j = 0; j < KK; ++j) {
    float best = -1e30f; int bi = 0;
    for (int nn = 0; nn < Nn; ++nn) {
      bool used = false;
      for (int q = 0; q < j; ++q) if (sel[q] == nn) { used = true; break; }
      if (used) continue;
      float v = ic[(size_t)nn * Mm];
      if (v > best) { best = v; bi = nn; }
    }
    sel[j] = bi; s += best;
  }
  int dk = (int)s;            // astype(int32) truncation
  if (dk < 1) dk = 1;
  dynk[t] = dk;

  for (int j = 0; j < KK; ++j) {
    float best = 1e30f; int bi = 0;
    for (int nn = 0; nn < Nn; ++nn) {
      bool used = false;
      for (int q = 0; q < j; ++q) if (sel[q] == nn) { used = true; break; }
      if (used) continue;
      float v = cc[(size_t)nn * Mm];
      if (v < best) { best = v; bi = nn; }
    }
    sel[j] = bi;
    idx_sel[t * KK + j] = bi;
  }
}

// ---------------- kernel 5: per-prediction match + losses -----------------
__global__ void k_loss(const float* __restrict__ logits,
                       const float* __restrict__ boxes,
                       const int*   __restrict__ gt_cls,
                       const float* __restrict__ gt_boxes,
                       const float* __restrict__ image_size,
                       const float* __restrict__ cost,
                       const float* __restrict__ validf,
                       const int*   __restrict__ idx_sel,
                       const int*   __restrict__ dynk,
                       const float* __restrict__ inter,
                       const float* __restrict__ rowsum,
                       const float* __restrict__ gmsum,
                       float* __restrict__ accum) {
  int tid = blockIdx.x * blockDim.x + threadIdx.x;
  if (tid >= Bb * Nn) return;
  int b = tid / Nn, n = tid % Nn;

  int msum = 0, first_m = 0;
  for (int m = 0; m < Mm; ++m) {
    int dk = dynk[b * Mm + m];
    const int* ix = idx_sel + (b * Mm + m) * KK;
    for (int j = 0; j < dk; ++j) {
      if (ix[j] == n) { if (msum == 0) first_m = m; ++msum; break; }
    }
  }
  bool validb = validf[tid] > 0.5f;
  int matched;
  bool fg;
  if (msum > 1) {
    float best = 1e38f; int bm = 0;
    for (int m = 0; m < Mm; ++m) {
      float v = cost[(size_t)tid * Mm + m];
      if (v < best) { best = v; bm = m; }
    }
    matched = bm; fg = validb;
  } else if (msum == 1) {
    matched = first_m; fg = validb;
  } else {
    matched = 0; fg = false;
  }
  if (!fg) return;

  int tcls = gt_cls[b * Mm + matched];

  float cls = 0.0f;
  const float* lg = logits + (size_t)tid * Cc;
  for (int c = 0; c < Cc; ++c) {
    float l = lg[c];
    float tg = (c == tcls) ? 1.0f : 0.0f;
    float ce = fmaxf(l, 0.0f) - l * tg + log1pf(__expf(-fabsf(l)));
    float p  = sigf(l);
    float pt = tg > 0.5f ? p : (1.0f - p);
    float w  = tg > 0.5f ? ALPHA_ : (1.0f - ALPHA_);
    cls += ce * (1.0f - pt) * (1.0f - pt) * w;
  }

  const float* bx  = boxes + (size_t)tid * 4;
  const float* img = image_size + b * 4;
  const float* g   = gt_boxes + (size_t)(b * Mm + matched) * 4;
  float nb0 = bx[0]/img[0], nb1 = bx[1]/img[1], nb2 = bx[2]/img[2], nb3 = bx[3]/img[3];
  float ng0 = g[0]/img[0],  ng1 = g[1]/img[1],  ng2 = g[2]/img[2],  ng3 = g[3]/img[3];
  float l1  = fabsf(nb0-ng0)+fabsf(nb1-ng1)+fabsf(nb2-ng2)+fabsf(nb3-ng3);
  float gio = 1.0f - d_giou(nb0,nb1,nb2,nb3, ng0,ng1,ng2,ng3);

  size_t o = (size_t)tid * Mm + matched;
  float mu = rowsum[o] + gmsum[b * Mm + matched] + EPSF;
  float ml = 1.0f - 2.0f * inter[o] / mu;

  atomicAdd(&accum[0], cls);
  atomicAdd(&accum[1], l1);
  atomicAdd(&accum[2], gio);
  atomicAdd(&accum[3], ml);
  atomicAdd(&accum[4], 1.0f);
}

// ---------------- kernel 6: finalize ----------------
__global__ void k_final(const float* __restrict__ accum, float* __restrict__ out) {
  float nsum = accum[4];
  out[0] = CLS_W  * accum[0] / nsum;
  out[1] = L1_W   * accum[1] / nsum;
  out[2] = GIOU_W * accum[2] / nsum;
  out[3] = MASK_W * accum[3] / nsum;
}

// ---------------- host launcher ----------------
extern "C" void kernel_launch(void* const* d_in, const int* in_sizes, int n_in,
                              void* d_out, int out_size, void* d_ws, size_t ws_size,
                              hipStream_t stream) {
  (void)in_sizes; (void)n_in; (void)out_size; (void)ws_size;
  const float* pred_logits = (const float*)d_in[0];
  const float* pred_boxes  = (const float*)d_in[1];
  const float* pred_masks  = (const float*)d_in[2];
  const int*   gt_classes  = (const int*)  d_in[3];
  const float* gt_boxes    = (const float*)d_in[4];
  const float* gt_masks    = (const float*)d_in[5];
  const float* image_size  = (const float*)d_in[6];
  float* out = (float*)d_out;

  char* w = (char*)d_ws;
  size_t off = 0;
  auto carve = [&](size_t bytes) { void* p = w + off; off = (off + bytes + 255) & ~(size_t)255; return p; };
  _Float16* gmh    = (_Float16*)carve((size_t)Bb*Mm*TTe*sizeof(_Float16));
  float*    gmsum  = (float*)   carve((size_t)Bb*Mm*sizeof(float));
  float*    interW = (float*)   carve((size_t)Bb*Nn*Mm*sizeof(float));
  float*    rowsum = (float*)   carve((size_t)Bb*Nn*Mm*sizeof(float));
  float*    costW  = (float*)   carve((size_t)Bb*Nn*Mm*sizeof(float));
  float*    iouvW  = (float*)   carve((size_t)Bb*Nn*Mm*sizeof(float));
  float*    validW = (float*)   carve((size_t)Bb*Nn*sizeof(float));
  int*      idxW   = (int*)     carve((size_t)Bb*Mm*KK*sizeof(int));
  int*      dynkW  = (int*)     carve((size_t)Bb*Mm*sizeof(int));
  float*    accum  = (float*)   carve(8*sizeof(float));

  k_init<<<1, 128, 0, stream>>>(gmsum, accum);
  k_crop<<<Bb*Mm, TTe, 0, stream>>>(gt_boxes, gt_masks, gmh, gmsum);
  k_wmma<<<Bb*NTIL*Mm, 32, 0, stream>>>(pred_masks, gt_classes, gmh, interW, rowsum);
  k_cost<<<(Bb*Nn + 127)/128, 128, 0, stream>>>(pred_logits, pred_boxes, gt_classes,
      gt_boxes, image_size, interW, rowsum, gmsum, costW, iouvW, validW);
  k_cols<<<1, 64, 0, stream>>>(iouvW, costW, idxW, dynkW);
  k_loss<<<(Bb*Nn + 127)/128, 128, 0, stream>>>(pred_logits, pred_boxes, gt_classes,
      gt_boxes, image_size, costW, validW, idxW, dynkW, interW, rowsum, gmsum, accum);
  k_final<<<1, 1, 0, stream>>>(accum, out);
}
